// StaticGNN_2748779069973
// MI455X (gfx1250) — matrix operand order repaired and checked by reading, not compiled, not run on hardware
//
#include <hip/hip_runtime.h>

// ---------------- sizes (compile-time, match reference) ----------------
#define NN      10000
#define EE      160000
#define RR      (EE + NN)      // edges + self loops = 170000
#define BB      16
#define F_NODE  42
#define F_EDGE  12
#define DD      256
#define HH      4
#define CC      64
#define LL      5
#define LH      (LL * HH)      // 20
#define NEG_SLOPE 0.2f

typedef __bf16 bf16_t;
typedef __attribute__((ext_vector_type(16))) __bf16 v16bf;
typedef __attribute__((ext_vector_type(8)))  __bf16 v8bf;
typedef __attribute__((ext_vector_type(8)))  float  v8f;

// ---------------- helpers ----------------
__device__ inline bf16_t f2bf(float f) {
    union { float f; unsigned u; } v; v.f = f;
    unsigned r = v.u + 0x7FFFu + ((v.u >> 16) & 1u);   // round-to-nearest-even
    union { unsigned short s; bf16_t b; } o;
    o.s = (unsigned short)(r >> 16);
    return o.b;
}
// order-preserving float <-> uint for atomicMax on signed floats
__device__ inline unsigned fenc(float x) {
    unsigned u = __float_as_uint(x);
    return (u & 0x80000000u) ? ~u : (u | 0x80000000u);
}
__device__ inline float fdec(unsigned u) {
    return __uint_as_float((u & 0x80000000u) ? (u & 0x7FFFFFFFu) : ~u);
}
#define ENC_NEG_INF 0x007FFFFFu   // fenc(-inf)

// ---------------- generic fill ----------------
__global__ void k_fill_u32(unsigned* p, unsigned val, int n) {
    int i = blockIdx.x * blockDim.x + threadIdx.x;
    if (i < n) p[i] = val;
}

// ---------------- f32 -> bf16 convert ----------------
__global__ void k_cvt_bf16(const float* __restrict__ src, bf16_t* __restrict__ dst, int n) {
    int i = blockIdx.x * blockDim.x + threadIdx.x;
    if (i < n) dst[i] = f2bf(src[i]);
}

// transpose + convert weight: W[k,n] (KxN f32 row-major) -> Wt[n,k] bf16
__global__ void k_transpose_cvt(const float* __restrict__ W, bf16_t* __restrict__ Wt) {
    int i = blockIdx.x * blockDim.x + threadIdx.x;   // n*256 + k
    if (i >= DD * DD) return;
    int n = i >> 8, k = i & 255;
    Wt[n * DD + k] = f2bf(W[k * DD + n]);
}

// ---------------- node encoder: h = relu(x@enc_w + b), also bf16 ----------------
__global__ __launch_bounds__(256) void k_encode_nodes(
    const float* __restrict__ x, const float* __restrict__ W, const float* __restrict__ b,
    float* __restrict__ h, bf16_t* __restrict__ hbf) {
    int n = blockIdx.x, c = threadIdx.x;
    float acc = b[c];
    const float* xr = x + n * F_NODE;
    #pragma unroll
    for (int k = 0; k < F_NODE; ++k) acc += xr[k] * W[k * DD + c];
    acc = fmaxf(acc, 0.f);
    h[(size_t)n * DD + c] = acc;
    hbf[(size_t)n * DD + c] = f2bf(acc);
}

// ---------------- edge encoder -> e_full[0..E), + esum/cnt for self loops -----
__global__ __launch_bounds__(256) void k_encode_edges(
    const float* __restrict__ ea, const float* __restrict__ W, const float* __restrict__ b,
    const int* __restrict__ dst0, float* __restrict__ e_full,
    float* __restrict__ esum, float* __restrict__ cnt) {
    int ed = blockIdx.x, c = threadIdx.x;
    float acc = b[c];
    const float* er = ea + ed * F_EDGE;
    #pragma unroll
    for (int k = 0; k < F_EDGE; ++k) acc += er[k] * W[k * DD + c];
    acc = fmaxf(acc, 0.f);
    e_full[(size_t)ed * DD + c] = acc;
    int d = dst0[ed];
    atomicAdd(&esum[(size_t)d * DD + c], acc);
    if (c == 0) atomicAdd(&cnt[d], 1.0f);
}

// self-loop rows: e_full[E+n] = esum[n] / max(cnt,1)
__global__ __launch_bounds__(256) void k_self_loops(
    const float* __restrict__ esum, const float* __restrict__ cnt, float* __restrict__ e_full) {
    int n = blockIdx.x, c = threadIdx.x;
    float cc = fmaxf(cnt[n], 1.0f);
    e_full[(size_t)(EE + n) * DD + c] = esum[(size_t)n * DD + c] / cc;
}

// Vall[k, l*H+h] = sum_c gat_ew[l][k, h*C+c] * att_edge[l,h,c]
__global__ void k_compute_vall(const float* __restrict__ gat_ew,
                               const float* __restrict__ att_edge, float* __restrict__ Vall) {
    int i = blockIdx.x * blockDim.x + threadIdx.x;   // k*LH + j
    if (i >= DD * LH) return;
    int k = i / LH, j = i % LH;
    int l = j >> 2, h = j & 3;
    const float* wr = gat_ew + (size_t)l * DD * DD + (size_t)k * DD + h * CC;
    const float* ar = att_edge + (size_t)(l * HH + h) * CC;
    float acc = 0.f;
    #pragma unroll
    for (int c = 0; c < CC; ++c) acc += wr[c] * ar[c];
    Vall[k * LH + j] = acc;
}

// a_e_all[row, j] = e_full[row,:] . Vall[:,j]  — one wave per row, single e_full pass
__global__ __launch_bounds__(256) void k_a_e_all(
    const float* __restrict__ e_full, const float* __restrict__ Vall, float* __restrict__ a_e) {
    int wid = (blockIdx.x * blockDim.x + threadIdx.x) >> 5;
    int lane = threadIdx.x & 31;
    if (wid >= RR) return;
    const float* er = e_full + (size_t)wid * DD;
    float acc[LH];
    #pragma unroll
    for (int j = 0; j < LH; ++j) acc[j] = 0.f;
    for (int kk = 0; kk < 8; ++kk) {
        int k = kk * 32 + lane;            // coalesced across lanes
        float ev = er[k];
        const float* vr = Vall + k * LH;
        #pragma unroll
        for (int j = 0; j < LH; ++j) acc[j] += ev * vr[j];
    }
    #pragma unroll
    for (int off = 16; off >= 1; off >>= 1)
        #pragma unroll
        for (int j = 0; j < LH; ++j) acc[j] += __shfl_xor(acc[j], off, 32);
    if (lane == 0) {
        float* o = a_e + (size_t)wid * LH;
        #pragma unroll
        for (int j = 0; j < LH; ++j) o[j] = acc[j];
    }
}

// ---------------- bf16 WMMA GEMM: C[M,256] = A[M,256] @ Bt^T, N=K=256 ----------
// Bt is the transposed weight (bf16, [N,K] row-major). mode: 0 plain, 1 +bias, 2 +bias+relu.
// blockDim=128 (4 waves); block -> 16 rows; wave w -> cols [w*64, w*64+64).
__global__ __launch_bounds__(128) void k_gemm_bf16_n256(
    const bf16_t* __restrict__ A, const bf16_t* __restrict__ Bt,
    float* __restrict__ Cmat, const float* __restrict__ bias, int mode) {
    const int K = DD;
    int wave = threadIdx.x >> 5;
    int lane = threadIdx.x & 31;
    int g = lane >> 4;          // half-wave group
    int ln = lane & 15;
    int rowBase = blockIdx.x * 16;
    int colBase = wave * 64;
    v8f c[4];
    #pragma unroll
    for (int t = 0; t < 4; ++t) c[t] = (v8f){0.f,0.f,0.f,0.f,0.f,0.f,0.f,0.f};
    const bf16_t* arow = A + (size_t)(rowBase + ln) * K;
    #pragma unroll
    for (int kt = 0; kt < 8; ++kt) {
        // A fragment (ISA 16-bit A 16x32 layout): two contiguous 8-element K runs
        int base0 = kt * 32 + g * 8;
        union { v16bf v; v8bf h[2]; } af;
        af.h[0] = *(const v8bf*)(arow + base0);
        af.h[1] = *(const v8bf*)(arow + base0 + 16);
        int kb = kt * 32 + g * 8;          // B: same two 8-element K runs as A
        #pragma unroll
        for (int t = 0; t < 4; ++t) {
            const bf16_t* bp = Bt + (size_t)(colBase + t * 16 + ln) * K + kb;
            union { v16bf v; v8bf h[2]; } bu;
            bu.h[0] = *(const v8bf*)bp;
            bu.h[1] = *(const v8bf*)(bp + 16);
            v16bf bfrag = bu.v;
            c[t] = __builtin_amdgcn_wmma_f32_16x16x32_bf16(
                       false, af.v, false, bfrag, (short)0, c[t], false, false);
        }
    }
    // epilogue: D element (i) -> M = i + 8*g, N = colBase + t*16 + ln
    #pragma unroll
    for (int t = 0; t < 4; ++t) {
        int n = colBase + t * 16 + ln;
        float bv = (mode >= 1) ? bias[n] : 0.f;
        #pragma unroll
        for (int i = 0; i < 8; ++i) {
            int m = rowBase + i + 8 * g;
            float v = c[t][i] + bv;
            if (mode == 2) v = fmaxf(v, 0.f);
            Cmat[(size_t)m * DD + n] = v;
        }
    }
}

// a_s[n,h], a_d[n,h] from xh
__global__ void k_as_ad(const float* __restrict__ xh, const float* __restrict__ att_s,
                        const float* __restrict__ att_d,
                        float* __restrict__ a_s, float* __restrict__ a_d) {
    int i = blockIdx.x * blockDim.x + threadIdx.x;
    if (i >= NN * HH) return;
    int n = i >> 2, h = i & 3;
    const float* xr = xh + (size_t)n * DD + h * CC;
    const float* as = att_s + h * CC;
    const float* ad = att_d + h * CC;
    float s = 0.f, d = 0.f;
    #pragma unroll
    for (int ci = 0; ci < CC; ++ci) { s += xr[ci] * as[ci]; d += xr[ci] * ad[ci]; }
    a_s[i] = s; a_d[i] = d;
}

// alpha = leaky_relu(a_s[src]+a_d[dst]+a_e); segment max into menc
__global__ void k_alpha_max(const float* __restrict__ a_s, const float* __restrict__ a_d,
                            const float* __restrict__ aeL,  // a_e_all + l*H
                            const int* __restrict__ src0, const int* __restrict__ dst0,
                            float* __restrict__ alpha, unsigned* __restrict__ menc) {
    long long i = (long long)blockIdx.x * blockDim.x + threadIdx.x;
    if (i >= (long long)RR * HH) return;
    int ed = (int)(i >> 2), h = (int)(i & 3);
    int s = (ed < EE) ? src0[ed] : ed - EE;
    int d = (ed < EE) ? dst0[ed] : ed - EE;
    float al = a_s[s * HH + h] + a_d[d * HH + h] + aeL[(size_t)ed * LH + h];
    al = (al > 0.f) ? al : NEG_SLOPE * al;
    alpha[(size_t)ed * HH + h] = al;
    atomicMax(&menc[d * HH + h], fenc(al));
}

// ex = exp(alpha - max); segment sum into den; alpha overwritten with ex
__global__ void k_exp_sum(const int* __restrict__ dst0, const unsigned* __restrict__ menc,
                          float* __restrict__ alpha, float* __restrict__ den) {
    long long i = (long long)blockIdx.x * blockDim.x + threadIdx.x;
    if (i >= (long long)RR * HH) return;
    int ed = (int)(i >> 2), h = (int)(i & 3);
    int d = (ed < EE) ? dst0[ed] : ed - EE;
    float ex = expf(alpha[(size_t)ed * HH + h] - fdec(menc[d * HH + h]));
    alpha[(size_t)ed * HH + h] = ex;
    atomicAdd(&den[d * HH + h], ex);
}

// out[dst] += xh[src] * attn   (block per edge row, thread per channel)
__global__ __launch_bounds__(256) void k_aggregate(
    const float* __restrict__ xh, const float* __restrict__ ex, const float* __restrict__ den,
    const int* __restrict__ src0, const int* __restrict__ dst0, float* __restrict__ outb) {
    int ed = blockIdx.x, c = threadIdx.x, h = c >> 6;
    int s = (ed < EE) ? src0[ed] : ed - EE;
    int d = (ed < EE) ? dst0[ed] : ed - EE;
    float w = ex[(size_t)ed * HH + h] / den[d * HH + h];
    atomicAdd(&outb[(size_t)d * DD + c], xh[(size_t)s * DD + c] * w);
}

// out += bias; LayerNorm; relu; h += out; emit bf16 h
__global__ __launch_bounds__(256) void k_ln_relu_res(
    const float* __restrict__ outb, const float* __restrict__ gb,
    const float* __restrict__ lg, const float* __restrict__ lb,
    float* __restrict__ h, bf16_t* __restrict__ hbf) {
    __shared__ float red[DD];
    int n = blockIdx.x, c = threadIdx.x;
    float v = outb[(size_t)n * DD + c] + gb[c];
    red[c] = v; __syncthreads();
    for (int s = 128; s > 0; s >>= 1) { if (c < s) red[c] += red[c + s]; __syncthreads(); }
    float mu = red[0] / (float)DD; __syncthreads();
    float dv = v - mu;
    red[c] = dv * dv; __syncthreads();
    for (int s = 128; s > 0; s >>= 1) { if (c < s) red[c] += red[c + s]; __syncthreads(); }
    float var = red[0] / (float)DD;
    float y = dv * rsqrtf(var + 1e-5f) * lg[c] + lb[c];
    y = fmaxf(y, 0.f);
    float hn = h[(size_t)n * DD + c] + y;
    h[(size_t)n * DD + c] = hn;
    hbf[(size_t)n * DD + c] = f2bf(hn);
}

// gate[n] = hidden[n,:] . gate_w2 + gate_b2
__global__ void k_gate2(const float* __restrict__ hid, const float* __restrict__ w2,
                        const float* __restrict__ b2, float* __restrict__ gate) {
    int n = blockIdx.x * blockDim.x + threadIdx.x;
    if (n >= NN) return;
    const float* hr = hid + (size_t)n * DD;
    float acc = b2[0];
    #pragma unroll 8
    for (int k = 0; k < DD; ++k) acc += hr[k] * w2[k];
    gate[n] = acc;
}

__global__ void k_gate_max(const float* __restrict__ gate, const int* __restrict__ batch,
                           unsigned* __restrict__ gmenc) {
    int n = blockIdx.x * blockDim.x + threadIdx.x;
    if (n >= NN) return;
    atomicMax(&gmenc[batch[n]], fenc(gate[n]));
}

__global__ void k_gate_expsum(const float* __restrict__ gate, const int* __restrict__ batch,
                              const unsigned* __restrict__ gmenc,
                              float* __restrict__ ge, float* __restrict__ gd) {
    int n = blockIdx.x * blockDim.x + threadIdx.x;
    if (n >= NN) return;
    int b = batch[n];
    float e = expf(gate[n] - fdec(gmenc[b]));
    ge[n] = e;
    atomicAdd(&gd[b], e);
}

__global__ __launch_bounds__(256) void k_pool(
    const float* __restrict__ h, const int* __restrict__ batch,
    const float* __restrict__ ge, const float* __restrict__ gd, float* __restrict__ gemb) {
    int n = blockIdx.x, c = threadIdx.x;
    int b = batch[n];
    float w = ge[n] / gd[b];
    atomicAdd(&gemb[(size_t)b * DD + c], h[(size_t)n * DD + c] * w);
}

// =======================================================================
extern "C" void kernel_launch(void* const* d_in, const int* in_sizes, int n_in,
                              void* d_out, int out_size, void* d_ws, size_t ws_size,
                              hipStream_t stream) {
    const float* x        = (const float*)d_in[0];
    const float* edgeattr = (const float*)d_in[1];
    const int*   eidx     = (const int*)d_in[2];
    const int*   batch    = (const int*)d_in[3];
    const float* enc_w    = (const float*)d_in[4];
    const float* enc_b    = (const float*)d_in[5];
    const float* eenc_w   = (const float*)d_in[6];
    const float* eenc_b   = (const float*)d_in[7];
    const float* gat_w    = (const float*)d_in[8];
    const float* att_src  = (const float*)d_in[9];
    const float* att_dst  = (const float*)d_in[10];
    const float* att_edge = (const float*)d_in[11];
    const float* gat_ew   = (const float*)d_in[12];
    const float* gat_b    = (const float*)d_in[13];
    const float* ln_g     = (const float*)d_in[14];
    const float* ln_b     = (const float*)d_in[15];
    const float* gate_w1  = (const float*)d_in[16];
    const float* gate_b1  = (const float*)d_in[17];
    const float* gate_w2  = (const float*)d_in[18];
    const float* gate_b2  = (const float*)d_in[19];
    const float* ro_w1    = (const float*)d_in[20];
    const float* ro_b1    = (const float*)d_in[21];
    const float* ro_w2    = (const float*)d_in[22];
    const float* ro_b2    = (const float*)d_in[23];
    const int* src0 = eidx;
    const int* dst0 = eidx + EE;

    // ---- workspace carve-out ----
    char* base = (char*)d_ws; size_t off = 0;
    auto alloc = [&](size_t bytes) -> void* {
        void* p = base + off; off = (off + bytes + 255) & ~(size_t)255; return p;
    };
    float*   e_full = (float*)alloc((size_t)RR * DD * 4);
    float*   esum   = (float*)alloc((size_t)NN * DD * 4);
    float*   cnt    = (float*)alloc((size_t)NN * 4);
    float*   a_e    = (float*)alloc((size_t)RR * LH * 4);
    float*   Vall   = (float*)alloc((size_t)DD * LH * 4);
    float*   h      = (float*)alloc((size_t)NN * DD * 4);
    bf16_t*  hbf    = (bf16_t*)alloc((size_t)NN * DD * 2);
    float*   xh     = (float*)alloc((size_t)NN * DD * 4);
    float*   a_s    = (float*)alloc((size_t)NN * HH * 4);
    float*   a_d    = (float*)alloc((size_t)NN * HH * 4);
    float*   alpha  = (float*)alloc((size_t)RR * HH * 4);
    unsigned* menc  = (unsigned*)alloc((size_t)NN * HH * 4);
    float*   den    = (float*)alloc((size_t)NN * HH * 4);
    float*   outb   = (float*)alloc((size_t)NN * DD * 4);
    bf16_t*  wt_gat = (bf16_t*)alloc((size_t)LL * DD * DD * 2);
    bf16_t*  wt_g1  = (bf16_t*)alloc((size_t)DD * DD * 2);
    bf16_t*  wt_r1  = (bf16_t*)alloc((size_t)DD * DD * 2);
    bf16_t*  wt_r2  = (bf16_t*)alloc((size_t)DD * DD * 2);
    float*   hidden = (float*)alloc((size_t)NN * DD * 4);
    float*   gate   = (float*)alloc((size_t)NN * 4);
    float*   ge     = (float*)alloc((size_t)NN * 4);
    unsigned* gmenc = (unsigned*)alloc((size_t)BB * 4);
    float*   gd     = (float*)alloc((size_t)BB * 4);
    float*   gemb   = (float*)alloc((size_t)BB * DD * 4);
    bf16_t*  gembbf = (bf16_t*)alloc((size_t)BB * DD * 2);
    float*   rohid  = (float*)alloc((size_t)BB * DD * 4);
    bf16_t*  rohbf  = (bf16_t*)alloc((size_t)BB * DD * 2);
    (void)ws_size; (void)in_sizes; (void)n_in; (void)out_size;

    auto g1 = [](int n) { return dim3((n + 255) / 256); };

    // ---- weight prep: transpose+bf16; attention-fold V_all ----
    for (int l = 0; l < LL; ++l)
        k_transpose_cvt<<<g1(DD * DD), 256, 0, stream>>>(gat_w + (size_t)l * DD * DD,
                                                         wt_gat + (size_t)l * DD * DD);
    k_transpose_cvt<<<g1(DD * DD), 256, 0, stream>>>(gate_w1, wt_g1);
    k_transpose_cvt<<<g1(DD * DD), 256, 0, stream>>>(ro_w1, wt_r1);
    k_transpose_cvt<<<g1(DD * DD), 256, 0, stream>>>(ro_w2, wt_r2);
    k_compute_vall<<<g1(DD * LH), 256, 0, stream>>>(gat_ew, att_edge, Vall);

    // ---- encoders + self-loop mean edge features ----
    k_encode_nodes<<<NN, 256, 0, stream>>>(x, enc_w, enc_b, h, hbf);
    k_fill_u32<<<g1(NN * DD), 256, 0, stream>>>((unsigned*)esum, 0u, NN * DD);
    k_fill_u32<<<g1(NN), 256, 0, stream>>>((unsigned*)cnt, 0u, NN);
    k_encode_edges<<<EE, 256, 0, stream>>>(edgeattr, eenc_w, eenc_b, dst0, e_full, esum, cnt);
    k_self_loops<<<NN, 256, 0, stream>>>(esum, cnt, e_full);

    // ---- one pass over e_full for all layers' edge-attention logits ----
    k_a_e_all<<<(RR * 32 + 255) / 256, 256, 0, stream>>>(e_full, Vall, a_e);

    // ---- GAT layers ----
    for (int l = 0; l < LL; ++l) {
        k_gemm_bf16_n256<<<NN / 16, 128, 0, stream>>>(hbf, wt_gat + (size_t)l * DD * DD,
                                                      xh, nullptr, 0);
        k_as_ad<<<g1(NN * HH), 256, 0, stream>>>(xh, att_src + (size_t)l * HH * CC,
                                                 att_dst + (size_t)l * HH * CC, a_s, a_d);
        k_fill_u32<<<g1(NN * HH), 256, 0, stream>>>(menc, ENC_NEG_INF, NN * HH);
        k_fill_u32<<<g1(NN * HH), 256, 0, stream>>>((unsigned*)den, 0u, NN * HH);
        k_alpha_max<<<g1(RR * HH), 256, 0, stream>>>(a_s, a_d, a_e + l * HH,
                                                     src0, dst0, alpha, menc);
        k_exp_sum<<<g1(RR * HH), 256, 0, stream>>>(dst0, menc, alpha, den);
        k_fill_u32<<<g1(NN * DD), 256, 0, stream>>>((unsigned*)outb, 0u, NN * DD);
        k_aggregate<<<RR, 256, 0, stream>>>(xh, alpha, den, src0, dst0, outb);
        k_ln_relu_res<<<NN, 256, 0, stream>>>(outb, gat_b + l * DD, ln_g + l * DD,
                                              ln_b + l * DD, h, hbf);
    }

    // ---- global attention pooling ----
    k_gemm_bf16_n256<<<NN / 16, 128, 0, stream>>>(hbf, wt_g1, hidden, gate_b1, 2);
    k_gate2<<<g1(NN), 256, 0, stream>>>(hidden, gate_w2, gate_b2, gate);
    k_fill_u32<<<1, 256, 0, stream>>>(gmenc, ENC_NEG_INF, BB);
    k_fill_u32<<<1, 256, 0, stream>>>((unsigned*)gd, 0u, BB);
    k_gate_max<<<g1(NN), 256, 0, stream>>>(gate, batch, gmenc);
    k_gate_expsum<<<g1(NN), 256, 0, stream>>>(gate, batch, gmenc, ge, gd);
    k_fill_u32<<<g1(BB * DD), 256, 0, stream>>>((unsigned*)gemb, 0u, BB * DD);
    k_pool<<<NN, 256, 0, stream>>>(h, batch, ge, gd, gemb);

    // ---- MLP readout (M=16, single WMMA row-tile each) ----
    k_cvt_bf16<<<g1(BB * DD), 256, 0, stream>>>(gemb, gembbf, BB * DD);
    k_gemm_bf16_n256<<<BB / 16, 128, 0, stream>>>(gembbf, wt_r1, rohid, ro_b1, 2);
    k_cvt_bf16<<<g1(BB * DD), 256, 0, stream>>>(rohid, rohbf, BB * DD);
    k_gemm_bf16_n256<<<BB / 16, 128, 0, stream>>>(rohbf, wt_r2, (float*)d_out, ro_b2, 1);
}
